// MultiheadAttention_25795573580066
// MI455X (gfx1250) — compile-verified, matrix operand order checked by reading, not run on hardware
//
#include <hip/hip_runtime.h>
#include <hip/hip_bf16.h>

// ---------- types ----------
typedef __bf16 bf16_t;
typedef __attribute__((ext_vector_type(16))) __bf16 v16bf;
typedef __attribute__((ext_vector_type(8)))  __bf16 v8bf;
typedef __attribute__((ext_vector_type(4)))  __bf16 v4bf;
typedef __attribute__((ext_vector_type(8)))  float  v8f;
typedef __attribute__((ext_vector_type(4)))  unsigned int u32x4;
typedef __attribute__((ext_vector_type(8)))  int i32x8;
typedef __attribute__((ext_vector_type(4)))  int i32x4;

// ---------- problem constants ----------
#define B_DIM  2
#define S_DIM  2048
#define E_DIM  1024
#define H_NUM  16
#define D_HEAD 64
#define M_ROWS (B_DIM * S_DIM)   // 4096

#define LOG2E 1.44269504088896340736f

// ---------- WMMA wrapper ----------
__device__ __forceinline__ v8f wmma_bf16f32(v16bf a, v16bf b, v8f c) {
  return __builtin_amdgcn_wmma_f32_16x16x32_bf16(
      false, a, false, b, (short)0, c, false, false);
}

// ---------- A/B fragment loader (16x32 bf16 row-major tile, stride ld) ----------
// lane L: row r = L&15; K-base kb = (L>=16)?8:0
// elems 0..7 = K kb..kb+7 ; elems 8..15 = K kb+16..kb+23
__device__ __forceinline__ v16bf load_frag(const bf16_t* tile, int ld) {
  const int lane = threadIdx.x & 31;
  const int r    = lane & 15;
  const int kb   = (lane & 16) >> 1;    // 0 or 8
  const bf16_t* p = tile + (size_t)r * ld + kb;
  v8bf lo = *reinterpret_cast<const v8bf*>(p);
  v8bf hi = *reinterpret_cast<const v8bf*>(p + 16);
  v16bf out;
#pragma unroll
  for (int i = 0; i < 8; ++i) { out[i] = lo[i]; out[i + 8] = hi[i]; }
  return out;
}

// ---------- Tensor Data Mover: 2D tile Global -> LDS ----------
// Builds the D# (group0 + group1) per CDNA5 ISA.
// data_size = 2 bytes (bf16). Padding inserted by the TDM itself:
//   pad_interval code c -> pad after 2^(c+1) DWORDs; pad_amount code a -> a+1 DWORDs.
__device__ __forceinline__ void tdm_load_2d(
    unsigned lds_addr, const void* gaddr,
    unsigned tensor_d0, unsigned tensor_d1,   // tensor extents (elements)
    unsigned tile_d0,   unsigned tile_d1,     // tile extents (elements)
    unsigned stride_d0,                       // dim0 stride (elements)
    unsigned pad_interval, unsigned pad_amount) {
  unsigned long long ga = (unsigned long long)(size_t)gaddr;
  u32x4 g0;
  g0.x = 1u;                                   // count=1: valid user descriptor
  g0.y = lds_addr;                             // LDS byte address
  g0.z = (unsigned)ga;                         // global_addr[31:0]
  g0.w = (unsigned)((ga >> 32) & 0x01ffffffull) | (2u << 30);  // [56:32] + type=2
  i32x8 g1;
  g1[0] = (int)((1u << 16)                     // data_size = 1 -> 2 bytes
              | (1u << 20)                     // pad_enable
              | (pad_interval << 22)
              | (pad_amount  << 25));
  g1[1] = (int)((tensor_d0 & 0xffffu) << 16);                  // tensor_dim0[15:0]
  g1[2] = (int)((tensor_d0 >> 16) | ((tensor_d1 & 0xffffu) << 16));
  g1[3] = (int)((tensor_d1 >> 16) | ((tile_d0 & 0xffffu) << 16));
  g1[4] = (int)(tile_d1 & 0xffffu);                            // tile_dim2 = 0
  g1[5] = (int)stride_d0;                                      // dim0 stride[31:0]
  g1[6] = 0;                                                   // stride hi, dim1 stride lo
  g1[7] = 0;
  i32x4 z4 = {0, 0, 0, 0};
#if __clang_major__ >= 23
  i32x8 z8 = {0, 0, 0, 0, 0, 0, 0, 0};
  __builtin_amdgcn_tensor_load_to_lds(g0, g1, z4, z4, z8, 0);
#else
  __builtin_amdgcn_tensor_load_to_lds(g0, g1, z4, z4, 0);
#endif
}

// ---------- fp32 -> bf16 convert (vectorized x4) ----------
__global__ void cvt_f32_bf16_kernel(const float4* __restrict__ x,
                                    v4bf* __restrict__ y, int n4) {
  int i = blockIdx.x * blockDim.x + threadIdx.x;
  if (i < n4) {
    float4 v = x[i];
    v4bf o;
    o[0] = (bf16_t)v.x; o[1] = (bf16_t)v.y; o[2] = (bf16_t)v.z; o[3] = (bf16_t)v.w;
    y[i] = o;
  }
}

// ---------- GEMM: Y = X(M,K) * W(N,K)^T + bias ----------
// Block tile 128(M) x 128(N); 8 waves arranged 2(M) x 4(N); wave tile 64x32.
// Per K=32 stage a wave does 8 WMMAs from 6 LDS fragments (B pair reused over
// 4 M-tiles). TDM double-buffered staging, LDS rows padded 64B->80B (ld=40,
// conflict-free ds_load_b128).
// mode 0: bf16 head-split [B,H,S,D] ; mode 1: bf16 transposed [B,H,D,S] ; mode 2: f32 flat
__global__ __launch_bounds__(256)
void gemm_bias_kernel(const bf16_t* __restrict__ X, const bf16_t* __restrict__ W,
                      const float* __restrict__ bias, void* __restrict__ out,
                      int M, int N, int K, int mode) {
  __shared__ bf16_t Xt[2][128][40];
  __shared__ bf16_t Wt[2][128][40];

  const int lane = threadIdx.x & 31;
  const int w    = threadIdx.x >> 5;       // 0..7
  const int wm   = w >> 2;                 // 0..1  (64-row slice)
  const int wn   = w & 3;                  // 0..3  (32-col slice)
  const int rowB = blockIdx.y * 128;
  const int colB = blockIdx.x * 128;
  const int steps = K / 32;

  if (threadIdx.x < 32) {   // wave 0 drives the TDM
    tdm_load_2d((unsigned)(size_t)&Xt[0][0][0], X + (size_t)rowB * K,
                K, M, 32, 128, K, /*pad: 16 dwords*/3, /*4 dwords*/3);
    tdm_load_2d((unsigned)(size_t)&Wt[0][0][0], W + (size_t)colB * K,
                K, N, 32, 128, K, 3, 3);
  }

  v8f acc[4][2];
#pragma unroll
  for (int mt = 0; mt < 4; ++mt)
#pragma unroll
    for (int nt = 0; nt < 2; ++nt)
#pragma unroll
      for (int i = 0; i < 8; ++i) acc[mt][nt][i] = 0.0f;

  for (int ki = 0; ki < steps; ++ki) {
    const int cb = ki & 1;
    if (threadIdx.x < 32) {
      if (ki + 1 < steps) {
        const int nb = cb ^ 1;
        const int k0n = (ki + 1) * 32;
        tdm_load_2d((unsigned)(size_t)&Xt[nb][0][0], X + (size_t)rowB * K + k0n,
                    K, M, 32, 128, K, 3, 3);
        tdm_load_2d((unsigned)(size_t)&Wt[nb][0][0], W + (size_t)colB * K + k0n,
                    K, N, 32, 128, K, 3, 3);
        __builtin_amdgcn_s_wait_tensorcnt(2);  // current buffer done, next in flight
      } else {
        __builtin_amdgcn_s_wait_tensorcnt(0);
      }
    }
    __syncthreads();

    const v16bf b0 = load_frag(&Wt[cb][wn * 32][0], 40);
    const v16bf b1 = load_frag(&Wt[cb][wn * 32 + 16][0], 40);
#pragma unroll
    for (int mt = 0; mt < 4; ++mt) {
      const v16bf a = load_frag(&Xt[cb][wm * 64 + mt * 16][0], 40);
      acc[mt][0] = wmma_bf16f32(a, b0, acc[mt][0]);
      acc[mt][1] = wmma_bf16f32(a, b1, acc[mt][1]);
    }

    __syncthreads();   // all reads done before buffer is re-filled
  }

  const int mlo  = (lane >> 4) << 3;
  const int ncol = lane & 15;
#pragma unroll
  for (int mt = 0; mt < 4; ++mt) {
#pragma unroll
    for (int nt = 0; nt < 2; ++nt) {
#pragma unroll
      for (int i = 0; i < 8; ++i) {
        const int gm = rowB + wm * 64 + mt * 16 + mlo + i;
        const int gn = colB + wn * 32 + nt * 16 + ncol;
        float v = acc[mt][nt][i] + bias[gn];
        if (mode == 2) {
          ((float*)out)[(size_t)gm * N + gn] = v;
        } else {
          const int bb = gm >> 11;
          const int ss = gm & (S_DIM - 1);
          const int hh = gn >> 6;
          const int dd = gn & (D_HEAD - 1);
          if (mode == 0) {
            ((bf16_t*)out)[(((size_t)(bb * H_NUM + hh) * S_DIM + ss) * D_HEAD) + dd] = (bf16_t)v;
          } else {
            ((bf16_t*)out)[(((size_t)(bb * H_NUM + hh) * D_HEAD + dd) * S_DIM) + ss] = (bf16_t)v;
          }
        }
      }
    }
  }
}

// ---------- Flash attention core ----------
// Qp,Kp: [B,H,S,D] bf16 ; Vt: [B,H,D,S] bf16 ; ctx: [B,S,E] bf16
// grid (S/128, B*H), 256 threads. K-block = 64 columns, TDM double-buffered
// K/V tiles shared by all 8 waves. LDS rows: 64 bf16 + 8 pad -> ld=72.
__global__ __launch_bounds__(256)
void attn_kernel(const bf16_t* __restrict__ Qp, const bf16_t* __restrict__ Kp,
                 const bf16_t* __restrict__ Vtg, bf16_t* __restrict__ ctx) {
  __shared__ bf16_t Kt[2][64][72];
  __shared__ bf16_t Vt[2][64][72];
  __shared__ bf16_t Plds[8][16][72];

  const int lane = threadIdx.x & 31;
  const int w    = threadIdx.x >> 5;
  const int bh   = blockIdx.y;
  const int bb   = bh >> 4;
  const int hh   = bh & (H_NUM - 1);
  const int q0   = blockIdx.x * 128 + w * 16;

  const bf16_t* Qbase = Qp  + ((size_t)bh * S_DIM + q0) * D_HEAD;
  const bf16_t* Kbase = Kp  + (size_t)bh * S_DIM * D_HEAD;
  const bf16_t* Vbase = Vtg + (size_t)bh * D_HEAD * S_DIM;

  const v16bf aQ0 = load_frag(Qbase, D_HEAD);       // d 0..31
  const v16bf aQ1 = load_frag(Qbase + 32, D_HEAD);  // d 32..63

  const int steps = S_DIM / 64;   // 32

  if (threadIdx.x < 32) {
    tdm_load_2d((unsigned)(size_t)&Kt[0][0][0], Kbase,
                D_HEAD, S_DIM, 64, 64, D_HEAD, /*32 dwords*/4, /*4 dwords*/3);
    tdm_load_2d((unsigned)(size_t)&Vt[0][0][0], Vbase,
                S_DIM, D_HEAD, 64, 64, S_DIM, 4, 3);
  }

  v8f acc[4];
#pragma unroll
  for (int t = 0; t < 4; ++t)
#pragma unroll
    for (int i = 0; i < 8; ++i) acc[t][i] = 0.0f;

  float mrow[8], lrow[8];
#pragma unroll
  for (int i = 0; i < 8; ++i) { mrow[i] = -3.0e38f; lrow[i] = 0.0f; }

  const int mlo  = (lane >> 4) << 3;
  const int ncol = lane & 15;
  const float sc = 0.125f;   // 1/sqrt(D)

  for (int kb = 0; kb < steps; ++kb) {
    const int cb = kb & 1;
    if (threadIdx.x < 32) {
      if (kb + 1 < steps) {
        const int nb  = cb ^ 1;
        const int k0n = (kb + 1) * 64;
        tdm_load_2d((unsigned)(size_t)&Kt[nb][0][0], Kbase + (size_t)k0n * D_HEAD,
                    D_HEAD, S_DIM, 64, 64, D_HEAD, 4, 3);
        tdm_load_2d((unsigned)(size_t)&Vt[nb][0][0], Vbase + k0n,
                    S_DIM, D_HEAD, 64, 64, S_DIM, 4, 3);
        __builtin_amdgcn_s_wait_tensorcnt(2);
      } else {
        __builtin_amdgcn_s_wait_tensorcnt(0);
      }
    }
    __syncthreads();

    // ---- scores: S[16x64] = Q(16x64) . K(64x64)^T ----
    v8f s[4];
#pragma unroll
    for (int nt = 0; nt < 4; ++nt) {
#pragma unroll
      for (int i = 0; i < 8; ++i) s[nt][i] = 0.0f;
      s[nt] = wmma_bf16f32(aQ0, load_frag(&Kt[cb][nt * 16][0], 72), s[nt]);
      s[nt] = wmma_bf16f32(aQ1, load_frag(&Kt[cb][nt * 16][32], 72), s[nt]);
    }

    // ---- online softmax per row (= mlo + i) ----
    float p[4][8];
#pragma unroll
    for (int i = 0; i < 8; ++i) {
      float r = s[0][i] * sc;
#pragma unroll
      for (int nt = 1; nt < 4; ++nt) r = fmaxf(r, s[nt][i] * sc);
      for (int mk = 1; mk < 16; mk <<= 1) r = fmaxf(r, __shfl_xor(r, mk, 32));
      const float Mn   = fmaxf(mrow[i], r);
      const float resc = exp2f((mrow[i] - Mn) * LOG2E);
      mrow[i] = Mn;
      float rs = 0.0f;
#pragma unroll
      for (int nt = 0; nt < 4; ++nt) {
        p[nt][i] = exp2f((s[nt][i] * sc - Mn) * LOG2E);
        rs += p[nt][i];
      }
      for (int mk = 1; mk < 16; mk <<= 1) rs += __shfl_xor(rs, mk, 32);
      lrow[i] = lrow[i] * resc + rs;
#pragma unroll
      for (int t = 0; t < 4; ++t) acc[t][i] *= resc;
    }

    // ---- stage P (bf16) to wave-private LDS ----
#pragma unroll
    for (int nt = 0; nt < 4; ++nt)
#pragma unroll
      for (int i = 0; i < 8; ++i)
        Plds[w][mlo + i][nt * 16 + ncol] = (bf16_t)p[nt][i];
    asm volatile("s_wait_dscnt 0" ::: "memory");   // wave-private, barrier-free

    // ---- out += P(16x64) . V(64x64) ----
    const v16bf aP0 = load_frag(&Plds[w][0][0], 72);
    const v16bf aP1 = load_frag(&Plds[w][0][32], 72);
#pragma unroll
    for (int dt = 0; dt < 4; ++dt) {
      acc[dt] = wmma_bf16f32(aP0, load_frag(&Vt[cb][dt * 16][0], 72), acc[dt]);
      acc[dt] = wmma_bf16f32(aP1, load_frag(&Vt[cb][dt * 16][32], 72), acc[dt]);
    }

    __syncthreads();   // all reads done before buffer refill
  }

  // ---- normalize + write ctx [B,S,E] ----
#pragma unroll
  for (int dt = 0; dt < 4; ++dt) {
#pragma unroll
    for (int i = 0; i < 8; ++i) {
      float o = acc[dt][i] / lrow[i];
      const int m = mlo + i;
      const size_t idx = ((size_t)bb * S_DIM + (q0 + m)) * E_DIM
                       + hh * D_HEAD + dt * 16 + ncol;
      ctx[idx] = (bf16_t)o;
    }
  }
}

// ---------- host-side launch ----------
extern "C" void kernel_launch(void* const* d_in, const int* in_sizes, int n_in,
                              void* d_out, int out_size, void* d_ws, size_t ws_size,
                              hipStream_t stream) {
  const float* q_in = (const float*)d_in[0];
  const float* k_in = (const float*)d_in[1];
  const float* v_in = (const float*)d_in[2];
  const float* Wq = (const float*)d_in[3];
  const float* bq = (const float*)d_in[4];
  const float* Wk = (const float*)d_in[5];
  const float* bk = (const float*)d_in[6];
  const float* Wv = (const float*)d_in[7];
  const float* bv = (const float*)d_in[8];
  const float* Wo = (const float*)d_in[9];
  const float* bo = (const float*)d_in[10];

  const size_t ACT_E = (size_t)M_ROWS * E_DIM;  // 4M elements
  const size_t W_E   = (size_t)E_DIM * E_DIM;   // 1M elements

  char* ws = (char*)d_ws;
  bf16_t* Xq   = (bf16_t*)(ws);
  bf16_t* Xk   = (bf16_t*)(ws + 1 * ACT_E * 2);
  bf16_t* Xv   = (bf16_t*)(ws + 2 * ACT_E * 2);
  bf16_t* Wq16 = (bf16_t*)(ws + 3 * ACT_E * 2);
  bf16_t* Wk16 = (bf16_t*)(ws + 3 * ACT_E * 2 + 1 * W_E * 2);
  bf16_t* Wv16 = (bf16_t*)(ws + 3 * ACT_E * 2 + 2 * W_E * 2);
  bf16_t* Wo16 = (bf16_t*)(ws + 3 * ACT_E * 2 + 3 * W_E * 2);
  bf16_t* Qp   = (bf16_t*)(ws + 3 * ACT_E * 2 + 4 * W_E * 2);
  bf16_t* Kp   = (bf16_t*)(ws + 4 * ACT_E * 2 + 4 * W_E * 2);
  bf16_t* Vt   = (bf16_t*)(ws + 5 * ACT_E * 2 + 4 * W_E * 2);
  bf16_t* ctx  = (bf16_t*)(ws + 6 * ACT_E * 2 + 4 * W_E * 2);

  // 1) fp32 -> bf16 conversions (x4 vectorized)
  {
    int n4 = (int)(ACT_E / 4), blocks = (n4 + 255) / 256;
    cvt_f32_bf16_kernel<<<blocks, 256, 0, stream>>>((const float4*)q_in, (v4bf*)Xq, n4);
    cvt_f32_bf16_kernel<<<blocks, 256, 0, stream>>>((const float4*)k_in, (v4bf*)Xk, n4);
    cvt_f32_bf16_kernel<<<blocks, 256, 0, stream>>>((const float4*)v_in, (v4bf*)Xv, n4);
    int nw4 = (int)(W_E / 4), wblocks = (nw4 + 255) / 256;
    cvt_f32_bf16_kernel<<<wblocks, 256, 0, stream>>>((const float4*)Wq, (v4bf*)Wq16, nw4);
    cvt_f32_bf16_kernel<<<wblocks, 256, 0, stream>>>((const float4*)Wk, (v4bf*)Wk16, nw4);
    cvt_f32_bf16_kernel<<<wblocks, 256, 0, stream>>>((const float4*)Wv, (v4bf*)Wv16, nw4);
    cvt_f32_bf16_kernel<<<wblocks, 256, 0, stream>>>((const float4*)Wo, (v4bf*)Wo16, nw4);
  }

  // 2) projections (M=4096, N=1024, K=1024); block tile 128x128
  dim3 ggrid(E_DIM / 128, M_ROWS / 128);
  gemm_bias_kernel<<<ggrid, 256, 0, stream>>>(Xq, Wq16, bq, Qp, M_ROWS, E_DIM, E_DIM, 0);
  gemm_bias_kernel<<<ggrid, 256, 0, stream>>>(Xk, Wk16, bk, Kp, M_ROWS, E_DIM, E_DIM, 0);
  gemm_bias_kernel<<<ggrid, 256, 0, stream>>>(Xv, Wv16, bv, Vt, M_ROWS, E_DIM, E_DIM, 1);

  // 3) attention core
  dim3 agrid(S_DIM / 128, B_DIM * H_NUM);
  attn_kernel<<<agrid, 256, 0, stream>>>(Qp, Kp, Vt, ctx);

  // 4) output projection -> f32 d_out
  gemm_bias_kernel<<<ggrid, 256, 0, stream>>>(ctx, Wo16, bo, d_out, M_ROWS, E_DIM, E_DIM, 2);
}
